// SelfAttentionLayer_71201967833933
// MI455X (gfx1250) — matrix operand order repaired and checked by reading, not compiled, not run on hardware
//
#include <hip/hip_runtime.h>

// MI455X / gfx1250 self-attention: bf16 WMMA (f32 accum) QKV projection +
// flash-attention with async-to-LDS double-buffered K/V staging.
// wave32; fragment layouts per CDNA5 ISA 7.12.2.

typedef __attribute__((ext_vector_type(8)))  __bf16 v8bf;
typedef __attribute__((ext_vector_type(16))) __bf16 v16bf;
typedef __attribute__((ext_vector_type(8)))  float  v8f;

namespace {
constexpr int kBatch = 2;
constexpr int kSeq   = 2048;
constexpr int kDim   = 1024;
constexpr int kHeads = 16;
constexpr int kDh    = 64;
constexpr int kRows  = kBatch * kSeq;   // 4096
constexpr int kCols  = kHeads * kDh;    // 1024
constexpr int kKT    = 64;              // keys staged per LDS tile
}

#if __has_builtin(__builtin_amdgcn_global_load_async_to_lds_b128)
#define ATTN_ASYNC_COPY 1
typedef int v4i __attribute__((vector_size(4 * sizeof(int))));
typedef __attribute__((address_space(1))) v4i gas_v4i;   // global
typedef __attribute__((address_space(3))) v4i las_v4i;   // LDS
#else
#define ATTN_ASYNC_COPY 0
#endif

union Frag16 { v16bf v; v8bf h[2]; };

__device__ __forceinline__ v8bf ld8(const __bf16* p) {
  return *reinterpret_cast<const v8bf*>(p);
}

__device__ __forceinline__ void async_wait0() {
#if __has_builtin(__builtin_amdgcn_s_wait_asynccnt)
  __builtin_amdgcn_s_wait_asynccnt(0);
#else
  asm volatile("s_wait_asynccnt 0x0" ::: "memory");
#endif
}

__device__ __forceinline__ float rmax16(float x) {
  x = fmaxf(x, __shfl_xor(x, 1));
  x = fmaxf(x, __shfl_xor(x, 2));
  x = fmaxf(x, __shfl_xor(x, 4));
  x = fmaxf(x, __shfl_xor(x, 8));
  return x;
}

__device__ __forceinline__ float rsum16(float x) {
  x += __shfl_xor(x, 1);
  x += __shfl_xor(x, 2);
  x += __shfl_xor(x, 4);
  x += __shfl_xor(x, 8);
  return x;
}

// ---------------------------------------------------------------- conversions
__global__ void cvt_x_kernel(const float* __restrict__ x, __bf16* __restrict__ xh) {
  size_t stride = (size_t)gridDim.x * blockDim.x;
  for (size_t i = (size_t)blockIdx.x * blockDim.x + threadIdx.x;
       i < (size_t)kRows * kDim; i += stride) {
    xh[i] = (__bf16)x[i];
  }
}

// Wt[z][n][k] = W_z[k][n]  (transposed so WMMA B-fragments are K-contiguous)
__global__ void cvt_w_kernel(const float* __restrict__ W0,
                             const float* __restrict__ W1,
                             const float* __restrict__ W2,
                             __bf16* __restrict__ Wt) {
  const float* src = (blockIdx.z == 0) ? W0 : (blockIdx.z == 1) ? W1 : W2;
  __bf16* dst = Wt + (size_t)blockIdx.z * kDim * kCols;
  size_t stride = (size_t)gridDim.x * blockDim.x;
  for (size_t i = (size_t)blockIdx.x * blockDim.x + threadIdx.x;
       i < (size_t)kDim * kCols; i += stride) {
    size_t k = i >> 10;
    size_t n = i & 1023;
    dst[n * kDim + k] = (__bf16)src[i];
  }
}

// ---------------------------------------------------------------- QKV GEMM
// C[4096,1024] = xh[4096,1024] @ W_z[1024,1024] via v_wmma_f32_16x16x32_bf16.
// z=0 -> Q (scaled by 1/sqrt(DH)) in [B,H,S,DH]; z=1 -> K in [B,H,S,DH];
// z=2 -> V transposed into [B,H,DH,S].
__global__ __launch_bounds__(256) void qkv_gemm_kernel(
    const __bf16* __restrict__ xh, const __bf16* __restrict__ Wt,
    __bf16* __restrict__ Qb, __bf16* __restrict__ Kb, __bf16* __restrict__ Vt) {
  const int lane = threadIdx.x & 31;
  const int wave = threadIdx.x >> 5;
  const int hl = lane >> 4;
  const int ln = lane & 15;
  const int z = blockIdx.z;
  const __bf16* wt = Wt + (size_t)z * kDim * kCols;
  const int rowbase = blockIdx.x * 128 + wave * 16;
  const int colbase = blockIdx.y * 64;

  v8f acc[4] = {};
  const __bf16* arow = xh + (size_t)(rowbase + ln) * kDim;

  for (int kb = 0; kb < kDim; kb += 32) {
    Frag16 a;
    a.h[0] = ld8(arow + kb + 16 * hl);
    a.h[1] = ld8(arow + kb + 16 * hl + 8);
#pragma unroll
    for (int g = 0; g < 4; ++g) {
      const __bf16* bp = wt + (size_t)(colbase + 16 * g + ln) * kDim + kb + 16 * hl;
      Frag16 bfr;
      bfr.h[0] = ld8(bp);
      bfr.h[1] = ld8(bp + 8);
      acc[g] = __builtin_amdgcn_wmma_f32_16x16x32_bf16(
          false, a.v, false, bfr.v, (short)0, acc[g], false, false);
    }
  }

#pragma unroll
  for (int g = 0; g < 4; ++g) {
#pragma unroll
    for (int v = 0; v < 8; ++v) {
      int gr = rowbase + v + 8 * hl;
      int gc = colbase + 16 * g + ln;
      int b  = gr >> 11;
      int s  = gr & (kSeq - 1);
      int h  = gc >> 6;
      int dh = gc & (kDh - 1);
      float val = acc[g][v];
      if (z == 0) {
        Qb[(((size_t)(b * kHeads + h) * kSeq + s) * kDh) + dh] = (__bf16)(val * 0.125f);
      } else if (z == 1) {
        Kb[(((size_t)(b * kHeads + h) * kSeq + s) * kDh) + dh] = (__bf16)val;
      } else {
        Vt[(((size_t)(b * kHeads + h) * kDh + dh) * kSeq) + s] = (__bf16)val;
      }
    }
  }
}

// ---------------------------------------------------------------- attention
// Block = 8 waves, one (b,h), 128 query rows. K/V tiles of 64 keys are staged
// into LDS once per block (async-to-LDS, double buffered); all waves build
// WMMA fragments from LDS.
__global__ __launch_bounds__(256) void attn_kernel(
    const __bf16* __restrict__ Qb, const __bf16* __restrict__ Kb,
    const __bf16* __restrict__ Vg, float* __restrict__ out) {
  __shared__ __align__(16) __bf16 Kl[2][kKT * kDh];    // [key][dh]  8 KB each
  __shared__ __align__(16) __bf16 Vl[2][kDh * kKT];    // [dh][key]  8 KB each
  __shared__ __align__(16) __bf16 pbuf[8][16 * 32];    // per-wave P tile

  const int tid  = threadIdx.x;
  const int lane = tid & 31;
  const int wave = tid >> 5;
  const int hl = lane >> 4;
  const int ln = lane & 15;
  const int bh = blockIdx.y;
  const int b  = bh >> 4;
  const int h  = bh & (kHeads - 1);
  const int qbase = blockIdx.x * 128 + wave * 16;

  const __bf16* Kbh = Kb + (size_t)bh * kSeq * kDh;   // [S][DH]
  const __bf16* Vbh = Vg + (size_t)bh * kDh * kSeq;   // [DH][S]

  // Q A-fragments (contraction over DH: 0..31 and 32..63); Q pre-scaled.
  const __bf16* qrow = Qb + ((size_t)bh * kSeq + (qbase + ln)) * kDh;
  Frag16 qa0, qa1;
  qa0.h[0] = ld8(qrow + 0 + 16 * hl);
  qa0.h[1] = ld8(qrow + 8 + 16 * hl);
  qa1.h[0] = ld8(qrow + 32 + 16 * hl);
  qa1.h[1] = ld8(qrow + 40 + 16 * hl);

  // Stage one 64-key K tile + V tile into LDS buffer `buf`.
  // 512 x 16B chunks each; 256 threads -> 2 chunks per array per thread.
  auto stage = [&](int buf, int kt) {
#pragma unroll
    for (int r = 0; r < 2; ++r) {
      int ci  = tid + r * 256;          // 0..511
      int row = ci >> 3;                // key (K) / dh (V)
      int c8  = ci & 7;                 // 16B chunk within 128B row
      const __bf16* gk = Kbh + (size_t)(kt + row) * kDh + c8 * 8;
      const __bf16* gv = Vbh + (size_t)row * kSeq + kt + c8 * 8;
      __bf16* lk = &Kl[buf][ci * 8];
      __bf16* lv = &Vl[buf][ci * 8];
#if ATTN_ASYNC_COPY
      __builtin_amdgcn_global_load_async_to_lds_b128(
          (gas_v4i*)(void*)gk, (las_v4i*)(void*)lk, 0, 0);
      __builtin_amdgcn_global_load_async_to_lds_b128(
          (gas_v4i*)(void*)gv, (las_v4i*)(void*)lv, 0, 0);
#else
      *reinterpret_cast<v8bf*>(lk) = ld8(gk);
      *reinterpret_cast<v8bf*>(lv) = ld8(gv);
#endif
    }
  };

  float m[8], l[8];
  v8f o[4] = {};
#pragma unroll
  for (int v = 0; v < 8; ++v) { m[v] = -3.0e38f; l[v] = 0.0f; }

  stage(0, 0);
#if ATTN_ASYNC_COPY
  async_wait0();
#endif
  __syncthreads();

  constexpr int kTiles = kSeq / kKT;    // 32
  for (int it = 0; it < kTiles; ++it) {
    if (it + 1 < kTiles) stage((it + 1) & 1, (it + 1) * kKT);

    const __bf16* Kt = Kl[it & 1];
    const __bf16* Vt = Vl[it & 1];

#pragma unroll
    for (int halfb = 0; halfb < 2; ++halfb) {   // two 32-key sub-blocks
      const int j0 = halfb * 32;

      // scores: keys j0..j0+15 -> c0, keys j0+16..j0+31 -> c1 (from LDS)
      v8f c0 = {}, c1 = {};
      {
        const __bf16* kp = Kt + (size_t)(j0 + ln) * kDh + 16 * hl;
        Frag16 b0, b1;
        b0.h[0] = ld8(kp);      b0.h[1] = ld8(kp + 8);    // dh 0..31
        b1.h[0] = ld8(kp + 32); b1.h[1] = ld8(kp + 40);   // dh 32..63
        c0 = __builtin_amdgcn_wmma_f32_16x16x32_bf16(false, qa0.v, false, b0.v, (short)0, c0, false, false);
        c0 = __builtin_amdgcn_wmma_f32_16x16x32_bf16(false, qa1.v, false, b1.v, (short)0, c0, false, false);
      }
      {
        const __bf16* kp = Kt + (size_t)(j0 + 16 + ln) * kDh + 16 * hl;
        Frag16 b0, b1;
        b0.h[0] = ld8(kp);      b0.h[1] = ld8(kp + 8);
        b1.h[0] = ld8(kp + 32); b1.h[1] = ld8(kp + 40);
        c1 = __builtin_amdgcn_wmma_f32_16x16x32_bf16(false, qa0.v, false, b0.v, (short)0, c1, false, false);
        c1 = __builtin_amdgcn_wmma_f32_16x16x32_bf16(false, qa1.v, false, b1.v, (short)0, c1, false, false);
      }

      // online softmax; row r = v + 8*hl is striped across 16 lanes (col = ln)
#pragma unroll
      for (int v = 0; v < 8; ++v) {
        float s0 = c0[v], s1 = c1[v];
        float rm = rmax16(fmaxf(s0, s1));
        float mn = fmaxf(m[v], rm);
        float al = __expf(m[v] - mn);
        float p0 = __expf(s0 - mn);
        float p1 = __expf(s1 - mn);
        l[v] = l[v] * al + rsum16(p0 + p1);
        m[v] = mn;
#pragma unroll
        for (int g = 0; g < 4; ++g) o[g][v] *= al;
        int r = v + 8 * hl;
        pbuf[wave][r * 32 + ln]      = (__bf16)p0;
        pbuf[wave][r * 32 + 16 + ln] = (__bf16)p1;
      }

      // P A-fragment (16 rows x 32 keys) from LDS, contiguous k order (k = 16*hl + s)
      Frag16 pf;
      const __bf16* pp = &pbuf[wave][ln * 32 + 16 * hl];
      pf.h[0] = ld8(pp);
      pf.h[1] = ld8(pp + 8);

      // O += P * V  (V tile in LDS is [dh][key], key-contiguous rows)
#pragma unroll
      for (int g = 0; g < 4; ++g) {
        const __bf16* vp = Vt + (size_t)(16 * g + ln) * kKT + j0 + 16 * hl;
        Frag16 vf;
        vf.h[0] = ld8(vp);
        vf.h[1] = ld8(vp + 8);
        o[g] = __builtin_amdgcn_wmma_f32_16x16x32_bf16(false, pf.v, false, vf.v, (short)0, o[g], false, false);
      }
    }

#if ATTN_ASYNC_COPY
    async_wait0();        // tile it+1 resident before anyone proceeds
#endif
    __syncthreads();      // all waves done with tile `it` before it's overwritten
  }

  // normalize and write out [B,S,H*DH] fp32
#pragma unroll
  for (int v = 0; v < 8; ++v) {
    float inv = 1.0f / l[v];
    int s = qbase + v + 8 * hl;
    float* orow = out + ((size_t)(b * kSeq + s)) * kCols + h * kDh;
#pragma unroll
    for (int g = 0; g < 4; ++g) orow[16 * g + ln] = o[g][v] * inv;
  }
}

// ---------------------------------------------------------------- launch
extern "C" void kernel_launch(void* const* d_in, const int* in_sizes, int n_in,
                              void* d_out, int out_size, void* d_ws, size_t ws_size,
                              hipStream_t stream) {
  (void)in_sizes; (void)n_in; (void)out_size; (void)ws_size;
  const float* x  = (const float*)d_in[0];
  const float* W0 = (const float*)d_in[1];
  const float* W1 = (const float*)d_in[2];
  const float* W2 = (const float*)d_in[3];
  float* out = (float*)d_out;

  char* ws = (char*)d_ws;
  __bf16* xh = (__bf16*)(ws);                          // 8 MiB  (4096x1024)
  __bf16* Wt = (__bf16*)(ws + ((size_t)8  << 20));     // 6 MiB  (3x1024x1024, transposed)
  __bf16* Qb = (__bf16*)(ws + ((size_t)14 << 20));     // 8 MiB  [B,H,S,DH] pre-scaled
  __bf16* Kb = (__bf16*)(ws + ((size_t)22 << 20));     // 8 MiB  [B,H,S,DH]
  __bf16* Vt = (__bf16*)(ws + ((size_t)30 << 20));     // 8 MiB  [B,H,DH,S]

  cvt_x_kernel<<<1024, 256, 0, stream>>>(x, xh);
  cvt_w_kernel<<<dim3(256, 1, 3), 256, 0, stream>>>(W0, W1, W2, Wt);
  qkv_gemm_kernel<<<dim3(32, 16, 3), 256, 0, stream>>>(xh, Wt, Qb, Kb, Vt);
  attn_kernel<<<dim3(16, 32), 256, 0, stream>>>(Qb, Kb, Vt, out);
}
